// RWKV5Block_79474074845663
// MI455X (gfx1250) — matrix operand order repaired and checked by reading, not compiled, not run on hardware
//
#include <hip/hip_runtime.h>
#include <hip/hip_bf16.h>

// ---------------- types ----------------
typedef unsigned short u16;
typedef __attribute__((ext_vector_type(16))) __bf16 v16bf;
typedef __attribute__((ext_vector_type(8)))  __bf16 v8bf;
typedef __attribute__((ext_vector_type(8)))  float  v8f;
typedef __attribute__((ext_vector_type(4)))  unsigned int u32x4;
typedef __attribute__((ext_vector_type(8)))  int i32x8;
typedef __attribute__((ext_vector_type(4)))  int i32x4;

#ifndef __has_builtin
#define __has_builtin(x) 0
#endif
#if __has_builtin(__builtin_amdgcn_tensor_load_to_lds) && __has_builtin(__builtin_amdgcn_s_wait_tensorcnt)
#define HAVE_TDM 1
#else
#define HAVE_TDM 0
#endif

__device__ __forceinline__ u16 f2bf(float f) {
    union { float f; unsigned u; } a; a.f = f;
    unsigned r = a.u + 0x7FFFu + ((a.u >> 16) & 1u);   // round-to-nearest-even
    return (u16)(r >> 16);
}

// ---------------- LayerNorm ----------------
__global__ __launch_bounds__(256) void ln_kernel(const float* __restrict__ x,
                                                 const float* __restrict__ g,
                                                 const float* __restrict__ b,
                                                 float* __restrict__ out, int C) {
    __shared__ float s1[256], s2[256];
    const size_t base = (size_t)blockIdx.x * C;
    float sum = 0.f, sq = 0.f;
    for (int c = threadIdx.x; c < C; c += 256) {
        float v = x[base + c]; sum += v; sq += v * v;
    }
    s1[threadIdx.x] = sum; s2[threadIdx.x] = sq; __syncthreads();
    for (int off = 128; off > 0; off >>= 1) {
        if (threadIdx.x < off) { s1[threadIdx.x] += s1[threadIdx.x + off];
                                 s2[threadIdx.x] += s2[threadIdx.x + off]; }
        __syncthreads();
    }
    const float mu  = s1[0] / C;
    const float var = s2[0] / C - mu * mu;
    const float rstd = rsqrtf(var + 1e-5f);
    for (int c = threadIdx.x; c < C; c += 256) {
        float v = x[base + c];
        out[base + c] = (v - mu) * rstd * g[c] + b[c];
    }
}

// ---------------- time-shift + token mix -> bf16 ----------------
__global__ __launch_bounds__(256) void mix4_kernel(const float* __restrict__ x,
    const float* __restrict__ mk, const float* __restrict__ mv,
    const float* __restrict__ mr, const float* __restrict__ mg,
    u16* __restrict__ ok, u16* __restrict__ ov, u16* __restrict__ orr, u16* __restrict__ og,
    int T, int C, long long total) {
    for (long long i = (long long)blockIdx.x * 256 + threadIdx.x; i < total;
         i += (long long)gridDim.x * 256) {
        const int c = (int)(i % C);
        const long long tc = i / C;
        const int t = (int)(tc % T);
        const float cur  = x[i];
        const float prev = (t > 0) ? x[i - C] : 0.f;
        float a;
        a = mk[c]; ok[i]  = f2bf(cur * a + prev * (1.f - a));
        a = mv[c]; ov[i]  = f2bf(cur * a + prev * (1.f - a));
        a = mr[c]; orr[i] = f2bf(cur * a + prev * (1.f - a));
        a = mg[c]; og[i]  = f2bf(cur * a + prev * (1.f - a));
    }
}

__global__ __launch_bounds__(256) void mix2_kernel(const float* __restrict__ x,
    const float* __restrict__ mk, const float* __restrict__ mr,
    u16* __restrict__ ok, u16* __restrict__ orr,
    int T, int C, long long total) {
    for (long long i = (long long)blockIdx.x * 256 + threadIdx.x; i < total;
         i += (long long)gridDim.x * 256) {
        const int c = (int)(i % C);
        const long long tc = i / C;
        const int t = (int)(tc % T);
        const float cur  = x[i];
        const float prev = (t > 0) ? x[i - C] : 0.f;
        float a;
        a = mk[c]; ok[i]  = f2bf(cur * a + prev * (1.f - a));
        a = mr[c]; orr[i] = f2bf(cur * a + prev * (1.f - a));
    }
}

// ---------------- f32 KxN -> bf16 NxK tiled transpose (weights) ----------------
__global__ __launch_bounds__(256) void cvt_t_bf16_kernel(const float* __restrict__ in,
                                                         u16* __restrict__ out,
                                                         int K, int N) {
    __shared__ float tile[32][33];
    const int kb = blockIdx.y * 32, nb = blockIdx.x * 32;
    const int tx = threadIdx.x & 31, ty = threadIdx.x >> 5;   // 32 x 8
    #pragma unroll
    for (int i = ty; i < 32; i += 8)
        tile[i][tx] = in[(size_t)(kb + i) * N + nb + tx];
    __syncthreads();
    #pragma unroll
    for (int i = ty; i < 32; i += 8)
        out[(size_t)(nb + i) * K + kb + tx] = f2bf(tile[tx][i]);
}

// ---------------- TDM tile loader (2D tile -> LDS, async, TENSORcnt) ----------
#if HAVE_TDM
__device__ __forceinline__ void tdm_load_2d(unsigned lds_byte_off, const u16* gptr,
                                            unsigned tile_cols, unsigned tile_rows,
                                            unsigned tensor_cols, unsigned tensor_rows,
                                            unsigned row_stride_elems) {
    const unsigned long long ga = (unsigned long long)(size_t)gptr;
    u32x4 g0;
    g0.x = 1u;                                                   // count=1 (valid, user mode)
    g0.y = lds_byte_off;                                         // lds_addr
    g0.z = (unsigned)ga;                                         // global_addr[31:0]
    g0.w = ((unsigned)(ga >> 32) & 0x01FFFFFFu) | 0x80000000u;   // addr[56:32] | type=2
    i32x8 g1;
    g1[0] = (int)(1u << 16);                                     // data_size = 1 -> 2 bytes
    g1[1] = (int)((tensor_cols & 0xFFFFu) << 16);                // tensor_dim0[15:0]
    g1[2] = (int)((tensor_cols >> 16) | ((tensor_rows & 0xFFFFu) << 16)); // dim0 hi | dim1 lo
    g1[3] = (int)((tensor_rows >> 16) | (tile_cols << 16));      // dim1 hi | tile_dim0
    g1[4] = (int)tile_rows;                                      // tile_dim1 (tile_dim2=0)
    g1[5] = (int)row_stride_elems;                               // tensor_dim0_stride[31:0]
    g1[6] = 0;                                                   // stride0[47:32], stride1 lo
    g1[7] = 0;                                                   // stride1 hi
    const i32x4 z4 = {0, 0, 0, 0};
    const i32x8 z8 = {0, 0, 0, 0, 0, 0, 0, 0};
    __builtin_amdgcn_tensor_load_to_lds(g0, g1, z4, z4, z8, 0);  // 6-arg toolchain form
}
#endif

// ---------------- WMMA inner tile (BK=32, 2x4 16x16 tiles per wave) ----------
__device__ __forceinline__ void compute_tile(const u16* sA, const u16* sB,
                                             int lrow, int lsel, int wm, int wn,
                                             v8f (&acc)[2][4]) {
    constexpr int BK = 32;
    // A frag: lane l -> row (l&15), K = (l>>4)*16 .. +15 contiguous
    v16bf afrag[2];
    #pragma unroll
    for (int i = 0; i < 2; ++i) {
        const u16* ap = sA + (wm * 32 + i * 16 + lrow) * BK + lsel * 16;
        union { v16bf v; v8bf h[2]; } u;
        u.h[0] = *(const v8bf*)(ap);
        u.h[1] = *(const v8bf*)(ap + 8);
        afrag[i] = u.v;
    }
    // B frag: lane l -> col (l&15), K = (l>>4)*16 .. +15 contiguous (B stored N-major)
    v16bf bfrag[4];
    #pragma unroll
    for (int j = 0; j < 4; ++j) {
        const u16* bp = sB + (wn * 64 + j * 16 + lrow) * BK + lsel * 16;
        bfrag[j] = *(const v16bf*)bp;
    }
    #pragma unroll
    for (int i = 0; i < 2; ++i)
        #pragma unroll
        for (int j = 0; j < 4; ++j)
            acc[i][j] = __builtin_amdgcn_wmma_f32_16x16x32_bf16(
                false, afrag[i], false, bfrag[j], (short)0, acc[i][j], false, false);
}

// ---------------- WMMA bf16 GEMM, TDM double-buffered, fused epilogues ------
enum { EPI_NONE = 0, EPI_SILU = 1, EPI_SIGMOID = 2, EPI_RELU2_BF16 = 3,
       EPI_ADD_RES = 4, EPI_FMA_RES = 5 };

template <int MODE>
__global__ __launch_bounds__(256) void gemm_bf16(
    const u16* __restrict__ Abf,   // M x K row-major bf16
    const u16* __restrict__ Btb,   // N x K row-major bf16 (pre-transposed weights)
    void* __restrict__ Cout,
    const float* __restrict__ Res, const float* __restrict__ Mul,
    int M, int N, int K) {
    constexpr int BM = 128, BN = 128, BK = 32;
    constexpr int ASZ = BM * BK, BSZ = BN * BK;
    __shared__ __align__(128) u16 smem[2 * (ASZ + BSZ)];   // double buffered A|B

    const int tid  = threadIdx.x;
    const int wid  = tid >> 5, lane = tid & 31;
    const int wm   = wid >> 1, wn = wid & 1;               // 4x2 wave grid
    const int lrow = lane & 15, lsel = lane >> 4;
    const int bm0  = blockIdx.y * BM;
    const int bn0  = blockIdx.x * BN;
    const int ntiles = K / BK;

    v8f acc[2][4];
    #pragma unroll
    for (int i = 0; i < 2; ++i)
        #pragma unroll
        for (int j = 0; j < 4; ++j) { v8f z = {}; acc[i][j] = z; }

#if HAVE_TDM
    const unsigned lds0 = (unsigned)(unsigned long long)(size_t)smem;
    if (wid == 0) {   // one wave drives the tensor DMA
        tdm_load_2d(lds0,           Abf + (size_t)bm0 * K, BK, BM, K, M, K);
        tdm_load_2d(lds0 + ASZ * 2, Btb + (size_t)bn0 * K, BK, BN, K, N, K);
    }
    for (int kt = 0; kt < ntiles; ++kt) {
        if (wid == 0) __builtin_amdgcn_s_wait_tensorcnt(0);   // tile kt resident
        __syncthreads();
        const int buf = kt & 1;
        if (wid == 0 && kt + 1 < ntiles) {                    // overlap next DMA w/ compute
            const unsigned dst = lds0 + (unsigned)((buf ^ 1) * (ASZ + BSZ) * 2);
            const int kn = (kt + 1) * BK;
            tdm_load_2d(dst,           Abf + (size_t)bm0 * K + kn, BK, BM, K, M, K);
            tdm_load_2d(dst + ASZ * 2, Btb + (size_t)bn0 * K + kn, BK, BN, K, N, K);
        }
        const u16* sA = smem + buf * (ASZ + BSZ);
        compute_tile(sA, sA + ASZ, lrow, lsel, wm, wn, acc);
        __syncthreads();
    }
#else
    // fallback: synchronous staging (A and Bt staged identically, contiguous)
    const int r0 = tid >> 2;            // 0..63 (and +64)
    const int c0 = (tid & 3) * 8;
    for (int k0 = 0; k0 < K; k0 += BK) {
        uint4 a0 = *(const uint4*)(Abf + (size_t)(bm0 + r0) * K + k0 + c0);
        uint4 a1 = *(const uint4*)(Abf + (size_t)(bm0 + r0 + 64) * K + k0 + c0);
        uint4 b0 = *(const uint4*)(Btb + (size_t)(bn0 + r0) * K + k0 + c0);
        uint4 b1 = *(const uint4*)(Btb + (size_t)(bn0 + r0 + 64) * K + k0 + c0);
        *(uint4*)(smem + r0 * BK + c0) = a0;
        *(uint4*)(smem + (r0 + 64) * BK + c0) = a1;
        *(uint4*)(smem + ASZ + r0 * BK + c0) = b0;
        *(uint4*)(smem + ASZ + (r0 + 64) * BK + c0) = b1;
        __syncthreads();
        compute_tile(smem, smem + ASZ, lrow, lsel, wm, wn, acc);
        __syncthreads();
    }
#endif

    // epilogue: D layout -> lane<16: M=r, N=lane; lane>=16: M=8+r, N=lane-16
    #pragma unroll
    for (int i = 0; i < 2; ++i) {
        #pragma unroll
        for (int j = 0; j < 4; ++j) {
            const int n     = bn0 + wn * 64 + j * 16 + lrow;
            const int mbase = bm0 + wm * 32 + i * 16 + lsel * 8;
            #pragma unroll
            for (int e = 0; e < 8; ++e) {
                const size_t idx = (size_t)(mbase + e) * N + n;
                float val = acc[i][j][e];
                if constexpr (MODE == EPI_SILU) {
                    ((float*)Cout)[idx] = val / (1.0f + __expf(-val));
                } else if constexpr (MODE == EPI_SIGMOID) {
                    ((float*)Cout)[idx] = 1.0f / (1.0f + __expf(-val));
                } else if constexpr (MODE == EPI_RELU2_BF16) {
                    float t = val > 0.f ? val : 0.f;
                    ((u16*)Cout)[idx] = f2bf(t * t);
                } else if constexpr (MODE == EPI_ADD_RES) {
                    ((float*)Cout)[idx] = Res[idx] + val;
                } else if constexpr (MODE == EPI_FMA_RES) {
                    ((float*)Cout)[idx] = Res[idx] + Mul[idx] * val;
                } else {
                    ((float*)Cout)[idx] = val;
                }
            }
        }
    }
}

// ---------------- WKV sequential recurrence ----------------
__global__ __launch_bounds__(64) void wkv_kernel(const float* __restrict__ r,
                                                 const float* __restrict__ k,
                                                 const float* __restrict__ v,
                                                 const float* __restrict__ td,
                                                 const float* __restrict__ tf,
                                                 float* __restrict__ out,
                                                 int T, int H) {
    constexpr int SS = 64;
    const int bh = blockIdx.x;
    const int b  = bh / H, h = bh % H;
    const int tid = threadIdx.x;
    __shared__ float wv_s[SS], uv_s[SS];
    __shared__ float kk[2][SS], rr[2][SS];
    wv_s[tid] = __expf(-__expf(td[h * SS + tid]));
    uv_s[tid] = tf[h * SS + tid];
    float s[SS];
    #pragma unroll
    for (int j = 0; j < SS; ++j) s[j] = 0.f;
    const int AA = H * SS;
    const size_t rowbase = (size_t)b * T * AA + (size_t)h * SS;
    __syncthreads();
    for (int t = 0; t < T; ++t) {
        const int buf = t & 1;
        const size_t base = rowbase + (size_t)t * AA;
        kk[buf][tid] = k[base + tid];
        rr[buf][tid] = r[base + tid];
        __syncthreads();
        const float vt = v[base + tid];
        float o = 0.f;
        #pragma unroll
        for (int j = 0; j < SS; ++j) {
            const float a = kk[buf][j] * vt;
            o += rr[buf][j] * (uv_s[j] * a + s[j]);
            s[j] = wv_s[j] * s[j] + a;
        }
        out[base + tid] = o;
    }
}

// ---------------- per-(b,t,h) groupnorm * gate -> bf16 ----------------
__global__ __launch_bounds__(64) void gnorm_kernel(const float* __restrict__ wkv,
                                                   const float* __restrict__ lnxg,
                                                   const float* __restrict__ lnxb,
                                                   const float* __restrict__ g,
                                                   u16* __restrict__ y, int H) {
    __shared__ float s1[64], s2[64];
    const int row = blockIdx.x / H, h = blockIdx.x % H, tid = threadIdx.x;
    const size_t base = (size_t)row * (H * 64) + h * 64 + tid;
    const float v = wkv[base];
    s1[tid] = v; s2[tid] = v * v; __syncthreads();
    for (int off = 32; off > 0; off >>= 1) {
        if (tid < off) { s1[tid] += s1[tid + off]; s2[tid] += s2[tid + off]; }
        __syncthreads();
    }
    const float mu  = s1[0] * (1.f / 64.f);
    const float var = s2[0] * (1.f / 64.f) - mu * mu;
    const float nv  = (v - mu) * rsqrtf(var + 1e-5f);
    const int c = h * 64 + tid;
    y[base] = f2bf((nv * lnxg[c] + lnxb[c]) * g[base]);
}

// ---------------- host-side launch ----------------
extern "C" void kernel_launch(void* const* d_in, const int* in_sizes, int n_in,
                              void* d_out, int out_size, void* d_ws, size_t ws_size,
                              hipStream_t stream) {
    constexpr int B = 4, T = 2048, C = 2048, H = 32, A = 2048, F = 7168;
    constexpr int NR = B * T;                      // 8192 rows
    constexpr long long NC = (long long)NR * C;    // activation elems (A == C)
    constexpr long long NF = (long long)NR * F;
    constexpr long long CA = (long long)C * A, CF = (long long)C * F,
                        FC = (long long)F * C, CC = (long long)C * C;

    const float* x      = (const float*)d_in[0];
    const float* ln0_g  = (const float*)d_in[1];
    const float* ln0_b  = (const float*)d_in[2];
    const float* ln1_g  = (const float*)d_in[3];
    const float* ln1_b  = (const float*)d_in[4];
    const float* ln2_g  = (const float*)d_in[5];
    const float* ln2_b  = (const float*)d_in[6];
    const float* mix_k  = (const float*)d_in[7];
    const float* mix_v  = (const float*)d_in[8];
    const float* mix_r  = (const float*)d_in[9];
    const float* mix_g  = (const float*)d_in[10];
    const float* Wr     = (const float*)d_in[11];
    const float* Wk     = (const float*)d_in[12];
    const float* Wv     = (const float*)d_in[13];
    const float* Wg     = (const float*)d_in[14];
    const float* td     = (const float*)d_in[15];
    const float* tf     = (const float*)d_in[16];
    const float* lnx_g  = (const float*)d_in[17];
    const float* lnx_b  = (const float*)d_in[18];
    const float* Wo     = (const float*)d_in[19];
    const float* fmix_k = (const float*)d_in[20];
    const float* fmix_r = (const float*)d_in[21];
    const float* fWk    = (const float*)d_in[22];
    const float* fWv    = (const float*)d_in[23];
    const float* fWr    = (const float*)d_in[24];
    float* out = (float*)d_out;

    // bump allocator over workspace (256B aligned offsets)
    char* ws = (char*)d_ws;
    size_t off = 0;
    auto alloc = [&](size_t bytes) -> void* {
        void* p = ws + off;
        off += (bytes + 255) & ~(size_t)255;
        return p;
    };
    float* x0   = (float*)alloc(NC * 4);
    float* x1   = (float*)alloc(NC * 4);
    float* xln  = (float*)alloc(NC * 4);
    u16*   xk   = (u16*)alloc(NC * 2);
    u16*   xv   = (u16*)alloc(NC * 2);
    u16*   xr   = (u16*)alloc(NC * 2);
    u16*   xg   = (u16*)alloc(NC * 2);
    u16*   wrb  = (u16*)alloc(CA * 2);    // A x C (transposed)
    u16*   wkb  = (u16*)alloc(CA * 2);
    u16*   wvb  = (u16*)alloc(CA * 2);
    u16*   wgb  = (u16*)alloc(CA * 2);
    u16*   wob  = (u16*)alloc(CA * 2);    // C x A (transposed)
    u16*   fwkb = (u16*)alloc(CF * 2);    // F x C (transposed)
    u16*   fwvb = (u16*)alloc(FC * 2);    // C x F (transposed)
    u16*   fwrb = (u16*)alloc(CC * 2);    // C x C (transposed)
    float* rb   = (float*)alloc(NC * 4);
    float* kb   = (float*)alloc(NC * 4);
    float* vb   = (float*)alloc(NC * 4);
    float* gb   = (float*)alloc(NC * 4);
    float* wkvo = (float*)alloc(NC * 4);
    u16*   yb   = (u16*)alloc(NC * 2);
    u16*   xk2  = (u16*)alloc(NC * 2);
    u16*   xr2  = (u16*)alloc(NC * 2);
    u16*   kf   = (u16*)alloc(NF * 2);
    float* rsig = (float*)alloc(NC * 4);
    (void)ws_size; (void)n_in; (void)in_sizes; (void)out_size;

    auto blocks = [](long long n) { return (unsigned)((n + 255) / 256); };

    // 1) x0 = ln0(x); xln = ln1(x0)
    ln_kernel<<<NR, 256, 0, stream>>>(x,  ln0_g, ln0_b, x0,  C);
    ln_kernel<<<NR, 256, 0, stream>>>(x0, ln1_g, ln1_b, xln, C);
    // 2) token mix (shifted) -> bf16
    mix4_kernel<<<blocks(NC), 256, 0, stream>>>(xln, mix_k, mix_v, mix_r, mix_g,
                                                xk, xv, xr, xg, T, C, NC);
    // 3) weights -> bf16, transposed to N x K for WMMA B operand
    cvt_t_bf16_kernel<<<dim3(A / 32, C / 32), 256, 0, stream>>>(Wr,  wrb,  C, A);
    cvt_t_bf16_kernel<<<dim3(A / 32, C / 32), 256, 0, stream>>>(Wk,  wkb,  C, A);
    cvt_t_bf16_kernel<<<dim3(A / 32, C / 32), 256, 0, stream>>>(Wv,  wvb,  C, A);
    cvt_t_bf16_kernel<<<dim3(A / 32, C / 32), 256, 0, stream>>>(Wg,  wgb,  C, A);
    cvt_t_bf16_kernel<<<dim3(C / 32, A / 32), 256, 0, stream>>>(Wo,  wob,  A, C);
    cvt_t_bf16_kernel<<<dim3(F / 32, C / 32), 256, 0, stream>>>(fWk, fwkb, C, F);
    cvt_t_bf16_kernel<<<dim3(C / 32, F / 32), 256, 0, stream>>>(fWv, fwvb, F, C);
    cvt_t_bf16_kernel<<<dim3(C / 32, C / 32), 256, 0, stream>>>(fWr, fwrb, C, C);
    // 4) QKVG projections (WMMA + TDM)
    {
        dim3 g(A / 128, NR / 128);
        gemm_bf16<EPI_NONE><<<g, 256, 0, stream>>>(xr, wrb, rb, nullptr, nullptr, NR, A, C);
        gemm_bf16<EPI_NONE><<<g, 256, 0, stream>>>(xk, wkb, kb, nullptr, nullptr, NR, A, C);
        gemm_bf16<EPI_NONE><<<g, 256, 0, stream>>>(xv, wvb, vb, nullptr, nullptr, NR, A, C);
        gemm_bf16<EPI_SILU><<<g, 256, 0, stream>>>(xg, wgb, gb, nullptr, nullptr, NR, A, C);
    }
    // 5) sequential WKV recurrence
    wkv_kernel<<<B * H, 64, 0, stream>>>(rb, kb, vb, td, tf, wkvo, T, H);
    // 6) per-head groupnorm, affine, * silu-gate -> bf16
    gnorm_kernel<<<NR * H, 64, 0, stream>>>(wkvo, lnx_g, lnx_b, gb, yb, H);
    // 7) output projection + residual: x1 = x0 + y @ Wo
    {
        dim3 g(C / 128, NR / 128);
        gemm_bf16<EPI_ADD_RES><<<g, 256, 0, stream>>>(yb, wob, x1, x0, nullptr, NR, C, A);
    }
    // 8) channel mix: xln = ln2(x1); mix -> xk2, xr2
    ln_kernel<<<NR, 256, 0, stream>>>(x1, ln2_g, ln2_b, xln, C);
    mix2_kernel<<<blocks(NC), 256, 0, stream>>>(xln, fmix_k, fmix_r, xk2, xr2, T, C, NC);
    // 9) k = relu(xk2 @ fWk)^2 -> bf16
    {
        dim3 g(F / 128, NR / 128);
        gemm_bf16<EPI_RELU2_BF16><<<g, 256, 0, stream>>>(xk2, fwkb, kf, nullptr, nullptr, NR, F, C);
    }
    // 10) rsig = sigmoid(xr2 @ fWr)
    {
        dim3 g(C / 128, NR / 128);
        gemm_bf16<EPI_SIGMOID><<<g, 256, 0, stream>>>(xr2, fwrb, rsig, nullptr, nullptr, NR, C, C);
    }
    // 11) out = x1 + rsig * (kf @ fWv)   (fused final epilogue, writes d_out)
    {
        dim3 g(C / 128, NR / 128);
        gemm_bf16<EPI_FMA_RES><<<g, 256, 0, stream>>>(kf, fwvb, out, x1, rsig, NR, C, F);
    }
}
